// aformer_4887672783302
// MI455X (gfx1250) — matrix operand order repaired and checked by reading, not compiled, not run on hardware
//
#include <hip/hip_runtime.h>
#include <hip/hip_bf16.h>

// ---------------------------------------------------------------------------
// Types for CDNA5 WMMA
// ---------------------------------------------------------------------------
typedef __attribute__((ext_vector_type(16))) __bf16        v16bf;
typedef __attribute__((ext_vector_type(8)))  float         v8f;
typedef __attribute__((ext_vector_type(8)))  unsigned short ush8;

union FragBF { v16bf v; ush8 h[2]; };
union AccF   { v8f v; float f[8]; };

__device__ __forceinline__ float bf2f(unsigned short h) {
    unsigned int u = ((unsigned int)h) << 16;
    float f; __builtin_memcpy(&f, &u, 4); return f;
}
__device__ __forceinline__ unsigned short f2bf(float f) {
    unsigned int u; __builtin_memcpy(&u, &f, 4);
    unsigned int r = u + 0x7FFFu + ((u >> 16) & 1u);
    return (unsigned short)(r >> 16);
}
__device__ __forceinline__ float gelu_t(float x) {
    const float c = 0.7978845608028654f;
    float t = tanhf(c * (x + 0.044715f * x * x * x));
    return 0.5f * x * (1.0f + t);
}
__device__ __forceinline__ float apply_epi(float x, int epi, float s, float a) {
    switch (epi) {
        case 1: return fmaxf(x, 0.0f);
        case 2: return gelu_t(x);
        case 3: return (x > 20.0f) ? x : log1pf(expf(x));
        case 4: return fmaxf(s * x, 0.0f) + a;
        default: return x;
    }
}

// ---------------------------------------------------------------------------
// Tensor Data Mover: 2-D tile DMA global -> LDS (CDNA5).  Descriptor per ISA §8.
// ---------------------------------------------------------------------------
#if __has_builtin(__builtin_amdgcn_tensor_load_to_lds)
#define USE_TDM 1
typedef __attribute__((ext_vector_type(4))) unsigned int u32x4;
typedef __attribute__((ext_vector_type(8))) int          i32x8;
typedef __attribute__((ext_vector_type(4))) int          i32x4;

// Load `rows` rows of `rowElems` bf16 elements, row stride `strideElems`,
// from gsrc into LDS at byte offset lds_byte_addr (contiguous row-major).
__device__ __forceinline__ void tdm_load_2d(const void* gsrc, unsigned lds_byte_addr,
                                            unsigned rows, unsigned rowElems,
                                            unsigned strideElems) {
    unsigned long long ga = (unsigned long long)gsrc;
    u32x4 g0;
    g0[0] = 1u;                                               // count=1 (valid), user mode
    g0[1] = lds_byte_addr;                                    // lds_addr
    g0[2] = (unsigned)(ga & 0xFFFFFFFFu);                     // global_addr[31:0]
    g0[3] = (unsigned)((ga >> 32) & 0x1FFFFFFu) | (2u << 30); // global_addr[56:32] | type=2
    i32x8 g1;
    g1[0] = 0x00010000;                                       // data_size=1 (2 bytes)
    g1[1] = (int)((strideElems & 0xFFFFu) << 16);             // tensor_dim0[15:0]
    g1[2] = (int)(((strideElems >> 16) & 0xFFFFu) |
                  ((rows & 0xFFFFu) << 16));                  // tensor_dim0 hi | tensor_dim1 lo
    g1[3] = (int)((rowElems & 0xFFFFu) << 16);                // tensor_dim1 hi=0 | tile_dim0
    g1[4] = (int)(rows & 0xFFFFu);                            // tile_dim1 | tile_dim2=0
    g1[5] = (int)strideElems;                                 // tensor_dim0_stride[31:0]
    g1[6] = 0;                                                // stride hi | dim1_stride lo
    g1[7] = 0;
    i32x4 z4 = {0, 0, 0, 0};
#if defined(__clang_major__) && (__clang_major__ >= 23)
    i32x8 z8 = {0, 0, 0, 0, 0, 0, 0, 0};
    __builtin_amdgcn_tensor_load_to_lds(g0, g1, z4, z4, z8, 0);
#else
    __builtin_amdgcn_tensor_load_to_lds(g0, g1, z4, z4, 0);
#endif
}
#endif

// All GEMMs use B in transposed layout:  Bt[n * K + k]  (bf16).
//
// ---------------------------------------------------------------------------
// FAST kernel: 64x64 tile / 256-thread block, 8 waves in 2(M)x4(N) grid,
// each wave owns a 32x16 subtile = 2 x v_wmma_f32_16x16x32_bf16 per k-step.
// Requires: M%64==0, N%64==0, K%32==0; CONV additionally Ci%32==0.
// Plain path stages tiles with the Tensor Data Mover when available.
// ---------------------------------------------------------------------------
template <int CONV>
__global__ __launch_bounds__(256) void wmma_gemm_fast(
    const unsigned short* __restrict__ A,
    const unsigned short* __restrict__ Bt,
    const float* __restrict__ bias,
    const unsigned short* __restrict__ resid,
    unsigned short* __restrict__ outBf,
    float* __restrict__ outF,
    int Mrows, int N, int K,
    int lda, int Ci, int dil, int padl, int Ltot,
    int epi, float escale, float eadd)
{
    __shared__ __align__(16) unsigned short lA[64 * 32];
    __shared__ __align__(16) unsigned short lB[64 * 32];

    int tid  = threadIdx.x;
    int lane = tid & 31;
    int wave = tid >> 5;
    int wm   = wave & 1;          // 2 M-subgroups of 32 rows
    int wn   = wave >> 1;         // 4 N-subgroups of 16 cols
    int half = lane >> 4;
    int l15  = lane & 15;

    int tileM = blockIdx.y * 64;
    int tileN = blockIdx.x * 64;

    int sr = tid >> 2;            // staging row 0..63
    int sc = (tid & 3) * 8;       // staging col 0,8,16,24
    (void)sr; (void)sc;

#if defined(USE_TDM)
    unsigned ldsA_off = (unsigned)(unsigned long long)(void*)lA;
    unsigned ldsB_off = (unsigned)(unsigned long long)(void*)lB;
#endif

    AccF acc0, acc1;
    #pragma unroll
    for (int i = 0; i < 8; ++i) { acc0.f[i] = 0.0f; acc1.f[i] = 0.0f; }

    for (int kt = 0; kt < K; kt += 32) {
        if constexpr (CONV != 0) {
            // ---- manual staging (conv halo may need zero-fill) ----
            ush8 av;
            int w  = kt / Ci;                        // uniform within tile (Ci%32==0)
            int cb = kt - w * Ci + sc;
            long srow = (long)(tileM + sr) + (long)w * dil - padl;
            if (srow >= 0 && srow < (long)Ltot) {
                av = *(const ush8*)&A[srow * lda + cb];
            } else {
                #pragma unroll
                for (int j = 0; j < 8; ++j) av[j] = 0;
            }
            *(ush8*)&lA[sr * 32 + sc] = av;
            ush8 bv = *(const ush8*)&Bt[(long)(tileN + sr) * K + kt + sc];
            *(ush8*)&lB[sr * 32 + sc] = bv;
            if (kt + 32 < K)
                __builtin_prefetch(&Bt[(long)(tileN + sr) * K + kt + 32 + sc], 0, 3);
        } else {
#if defined(USE_TDM)
            // ---- Tensor Data Mover staging (wave 0 issues both tiles) ----
            if (wave == 0) {
                tdm_load_2d(&A[(long)tileM * lda + kt], ldsA_off, 64u, 32u, (unsigned)lda);
                tdm_load_2d(&Bt[(long)tileN * K + kt], ldsB_off, 64u, 32u, (unsigned)K);
                __builtin_amdgcn_s_wait_tensorcnt(0);
            }
#else
            ush8 av = *(const ush8*)&A[(long)(tileM + sr) * lda + kt + sc];
            *(ush8*)&lA[sr * 32 + sc] = av;
            ush8 bv = *(const ush8*)&Bt[(long)(tileN + sr) * K + kt + sc];
            *(ush8*)&lB[sr * 32 + sc] = bv;
            if (kt + 32 < K)
                __builtin_prefetch(&Bt[(long)(tileN + sr) * K + kt + 32 + sc], 0, 3);
#endif
        }
        __syncthreads();

        // ---- fragments ----
        const ush8* ar0 = (const ush8*)&lA[(wm * 32 + l15) * 32];
        const ush8* ar1 = (const ush8*)&lA[(wm * 32 + 16 + l15) * 32];
        FragBF a0; a0.h[0] = ar0[half];     a0.h[1] = ar0[2 + half];
        FragBF a1; a1.h[0] = ar1[half];     a1.h[1] = ar1[2 + half];
        const ush8* br  = (const ush8*)&lB[(wn * 16 + l15) * 32];
        FragBF bf_; bf_.h[0] = br[half]; bf_.h[1] = br[2 + half];

        acc0.v = __builtin_amdgcn_wmma_f32_16x16x32_bf16(
            false, a0.v, false, bf_.v, (short)0, acc0.v, false, false);
        acc1.v = __builtin_amdgcn_wmma_f32_16x16x32_bf16(
            false, a1.v, false, bf_.v, (short)0, acc1.v, false, false);
        __syncthreads();
    }

    // ---- epilogue (no guards: tile-aligned) ----
    int col = tileN + wn * 16 + l15;
    float bvv = bias ? bias[col] : 0.0f;
    #pragma unroll
    for (int s = 0; s < 2; ++s) {
        AccF& ac = s ? acc1 : acc0;
        #pragma unroll
        for (int r = 0; r < 8; ++r) {
            int rowl = tileM + wm * 32 + s * 16 + r + 8 * half;
            long oidx = (long)rowl * N + col;
            float x = ac.f[r] + bvv;
            if (resid) x += bf2f(resid[oidx]);
            x = apply_epi(x, epi, escale, eadd);
            if (outF)  outF[oidx]  = x;
            if (outBf) outBf[oidx] = f2bf(x);
        }
    }
}

// ---------------------------------------------------------------------------
// GUARDED generic kernel, templated on addressing mode:
//   MODE 0: plain  A(m,k) = A[(row0+m)*lda + k]
//   MODE 1: conv   A(m,kk) with kk = w*Ci + ci, Ci = 1<<ciShift (power of 2)
//   MODE 2: transposed A(m,k) = A[k*lda + m]; output transposed bf16 only
// Staging is branchless: indices clamped in-range, value masked by validity.
// ---------------------------------------------------------------------------
#define TM 32
#define TN 64
#define TK 32

template <int MODE>
__global__ __launch_bounds__(256) void wmma_gemm_gen(
    const unsigned short* __restrict__ A,
    const unsigned short* __restrict__ Bt,
    const float* __restrict__ bias,
    const unsigned short* __restrict__ resid,
    unsigned short* __restrict__ outBf,
    float* __restrict__ outF,
    int Mrows, int N, int K,
    int lda, int ciShift, int dil, int padl,
    int row0, int Ltot,
    int epi, float escale, float eadd)
{
    __shared__ __align__(16) unsigned short lA[TM * TK];
    __shared__ __align__(16) unsigned short lB[TN * TK];

    int tid  = threadIdx.x;
    int lane = tid & 31;
    int wave = tid >> 5;
    int wm   = wave & 1;
    int wn   = wave >> 1;
    int half = lane >> 4;
    int l15  = lane & 15;

    int tileM = blockIdx.y * TM;
    int tileN = blockIdx.x * TN;

    AccF acc;
    #pragma unroll
    for (int i = 0; i < 8; ++i) acc.f[i] = 0.0f;

    for (int kt = 0; kt < K; kt += TK) {
        #pragma unroll
        for (int it = 0; it < (TM * TK) / 256; ++it) {
            int i  = tid + it * 256;
            int m  = i >> 5;
            int kk = i & 31;
            int gk = kt + kk;
            int gkc = gk < K ? gk : K - 1;
            unsigned short v;
            bool ok;
            if constexpr (MODE == 1) {
                int w  = gkc >> ciShift;
                int ci = gkc & ((1 << ciShift) - 1);
                long srow  = (long)(row0 + tileM + m) + (long)w * dil - padl;
                long srowc = srow < 0 ? 0 : (srow >= (long)Ltot ? (long)Ltot - 1 : srow);
                v  = A[srowc * lda + ci];
                ok = (gk < K) & (srow >= 0) & (srow < (long)Ltot) & ((tileM + m) < Mrows);
            } else if constexpr (MODE == 2) {
                int mg  = tileM + m;
                int mgc = mg < Mrows ? mg : Mrows - 1;
                v  = A[(long)gkc * lda + mgc];
                ok = (gk < K) & (mg < Mrows);
            } else {
                int mg  = tileM + m;
                int mgc = mg < Mrows ? mg : Mrows - 1;
                v  = A[(long)(row0 + mgc) * lda + gkc];
                ok = (gk < K) & (mg < Mrows);
            }
            lA[m * TK + kk] = ok ? v : (unsigned short)0;
        }
        #pragma unroll
        for (int it = 0; it < (TN * TK) / 256; ++it) {
            int i  = tid + it * 256;
            int n  = i >> 5;
            int kk = i & 31;
            int gk = kt + kk;
            int gn = tileN + n;
            int gkc = gk < K ? gk : K - 1;
            int gnc = gn < N ? gn : N - 1;
            unsigned short v = Bt[(long)gnc * K + gkc];
            bool ok = (gk < K) & (gn < N);
            lB[n * TK + kk] = ok ? v : (unsigned short)0;
        }
        __syncthreads();

        const ush8* arow = (const ush8*)&lA[(wm * 16 + l15) * TK];
        FragBF af; af.h[0] = arow[half];     af.h[1] = arow[2 + half];
        const ush8* brow = (const ush8*)&lB[(wn * 16 + l15) * TK];
        FragBF bfm; bfm.h[0] = brow[half]; bfm.h[1] = brow[2 + half];

        acc.v = __builtin_amdgcn_wmma_f32_16x16x32_bf16(
            false, af.v, false, bfm.v, (short)0, acc.v, false, false);
        __syncthreads();
    }

    int col = tileN + wn * 16 + l15;
    if (col < N) {
        float bv = bias ? bias[col] : 0.0f;
        #pragma unroll
        for (int r = 0; r < 8; ++r) {
            int rowl = tileM + wm * 16 + r + 8 * half;
            if (rowl < Mrows) {
                float x = acc.f[r] + bv;
                if constexpr (MODE == 2) {
                    x = apply_epi(x, epi, escale, eadd);
                    if (outBf) outBf[(long)col * Mrows + rowl] = f2bf(x);
                } else {
                    long oidx = (long)rowl * N + col;
                    if (resid) x += bf2f(resid[oidx]);
                    x = apply_epi(x, epi, escale, eadd);
                    if (outF)  outF[oidx]  = x;
                    if (outBf) outBf[oidx] = f2bf(x);
                }
            }
        }
    }
}

// ---------------------------------------------------------------------------
// Elementwise / reduction kernels
// ---------------------------------------------------------------------------
__global__ void f2bf_kernel(const float* __restrict__ in, unsigned short* __restrict__ out, long n) {
    long i = (long)blockIdx.x * 256 + threadIdx.x;
    if (i < n) out[i] = f2bf(in[i]);
}

// out[c * R + r] = in[r * C + c]   (weights -> Bt layout, R = K, C = N)
__global__ void transpose_f2bf_kernel(const float* __restrict__ in, unsigned short* __restrict__ out,
                                      int R, int C) {
    long i = (long)blockIdx.x * 256 + threadIdx.x;
    if (i >= (long)R * C) return;
    int r = (int)(i / C), c = (int)(i % C);
    out[(long)c * R + r] = f2bf(in[i]);
}

__global__ void bn_gelu_kernel(const unsigned short* __restrict__ x,
                               const float* __restrict__ bb, const float* __restrict__ gg,
                               const float* __restrict__ mm, const float* __restrict__ vv,
                               unsigned short* __restrict__ y, long n, int C) {
    long i = (long)blockIdx.x * 256 + threadIdx.x;
    if (i >= n) return;
    int c = (int)(i % C);
    float xv = bf2f(x[i]);
    float norm = (xv - mm[c]) * (gg[c] * rsqrtf(vv[c] + 1e-5f)) + bb[c];
    y[i] = f2bf(gelu_t(norm));
}

__global__ void smpool_kernel(const unsigned short* __restrict__ x,
                              const unsigned short* __restrict__ e,
                              unsigned short* __restrict__ y, long nout, int C, int pool) {
    long i = (long)blockIdx.x * 256 + threadIdx.x;
    if (i >= nout) return;
    long lo = i / C;
    int c = (int)(i % C);
    float ev[4]; float mx = -1e30f;
    for (int p = 0; p < pool; ++p) {
        ev[p] = bf2f(e[(lo * pool + p) * C + c]);
        mx = fmaxf(mx, ev[p]);
    }
    float den = 0.0f;
    for (int p = 0; p < pool; ++p) { ev[p] = expf(ev[p] - mx); den += ev[p]; }
    float s = 0.0f;
    for (int p = 0; p < pool; ++p) s += bf2f(x[(lo * pool + p) * C + c]) * ev[p];
    y[i] = f2bf(s / den);
}

__global__ __launch_bounds__(256) void layernorm_kernel(const unsigned short* __restrict__ x,
                                                        const float* __restrict__ g,
                                                        const float* __restrict__ b,
                                                        unsigned short* __restrict__ y,
                                                        int C, float eps) {
    __shared__ float s1[256], s2[256];
    int row = blockIdx.x;
    const unsigned short* xr = x + (long)row * C;
    float a = 0.0f, q = 0.0f;
    for (int c = threadIdx.x; c < C; c += 256) {
        float v = bf2f(xr[c]); a += v; q += v * v;
    }
    s1[threadIdx.x] = a; s2[threadIdx.x] = q;
    __syncthreads();
    for (int o = 128; o > 0; o >>= 1) {
        if (threadIdx.x < o) { s1[threadIdx.x] += s1[threadIdx.x + o]; s2[threadIdx.x] += s2[threadIdx.x + o]; }
        __syncthreads();
    }
    float mu = s1[0] / C;
    float var = s2[0] / C - mu * mu;
    float inv = rsqrtf(var + eps);
    for (int c = threadIdx.x; c < C; c += 256) {
        float v = bf2f(xr[c]);
        y[(long)row * C + c] = f2bf((v - mu) * inv * g[c] + b[c]);
    }
}

__global__ void concat_kernel(const unsigned short* __restrict__ xs,
                              const unsigned short* __restrict__ xa,
                              unsigned short* __restrict__ o, int L) {
    long i = (long)blockIdx.x * 256 + threadIdx.x;
    if (i >= (long)L * 1600) return;
    int l = (int)(i / 1600), c = (int)(i % 1600);
    o[i] = (c < 1536) ? xs[(long)l * 1536 + c] : xa[(long)l * 64 + (c - 1536)];
}

__global__ void addpe_kernel(unsigned short* __restrict__ x, int L, int D) {
    long i = (long)blockIdx.x * 256 + threadIdx.x;
    if (i >= (long)L * D) return;
    int l = (int)(i / D), c = (int)(i % D);
    float ang = (float)l * powf(10000.0f, -2.0f * (float)(c / 2) / (float)D);
    float pe = (c & 1) ? cosf(ang) : sinf(ang);
    x[i] = f2bf(bf2f(x[i]) + pe);
}

// q/k/v: [L,1600] f32 -> per-head bf16.  mode 1: RoPE, [h][l][400]
//                                        mode 0: plain, [h][l][400]
//                                        mode 2: plain, transposed [h][400][L] (Bt layout)
__global__ void rope_split_kernel(const float* __restrict__ src,
                                  unsigned short* __restrict__ dst, int L, int mode) {
    long i = (long)blockIdx.x * 256 + threadIdx.x;
    if (i >= (long)L * 1600) return;
    int l = (int)(i / 1600), c = (int)(i % 1600);
    int h = c / 400, d = c % 400;
    float x = src[i];
    float y = x;
    if (mode == 1) {
        int dd = d % 200;
        float inv = powf(10000.0f, -2.0f * (float)dd / 400.0f);
        float f = (float)l * inv;
        float s = sinf(f), cs = cosf(f);
        float other = (d < 200) ? src[i + 200] : src[i - 200];
        y = (d < 200) ? (x * cs - other * s) : (x * cs + other * s);
    }
    if (mode == 2)
        dst[(long)h * 400 * L + (long)d * L + l] = f2bf(y);
    else
        dst[(long)h * L * 400 + (long)l * 400 + d] = f2bf(y);
}

// ks[h][m] = sum_l kp[h][l][m]
__global__ void ksum_kernel(const float* __restrict__ kp, float* __restrict__ ks, int L) {
    int i = blockIdx.x * 256 + threadIdx.x;
    if (i >= 4 * 256) return;
    int h = i >> 8, m = i & 255;
    const float* base = kp + (long)h * L * 256 + m;
    float s = 0.0f;
    for (int l = 0; l < L; ++l) s += base[(long)l * 256];
    ks[i] = s;
}

// z[l][h] = sum_m qp[h][l][m] * ks[h][m]
__global__ void zden_kernel(const float* __restrict__ qp, const float* __restrict__ ks,
                            float* __restrict__ z, int L) {
    int i = blockIdx.x * 256 + threadIdx.x;
    if (i >= L * 4) return;
    int l = i >> 2, h = i & 3;
    const float* qb = qp + (long)h * L * 256 + (long)l * 256;
    const float* kb = ks + h * 256;
    float s = 0.0f;
    for (int m = 0; m < 256; ++m) s += qb[m] * kb[m];
    z[i] = s;
}

// o[l][h*400+d] = num[h][l][d] / z[l][h]
__global__ void combine_kernel(const float* __restrict__ num, const float* __restrict__ z,
                               unsigned short* __restrict__ o, int L) {
    long i = (long)blockIdx.x * 256 + threadIdx.x;
    if (i >= (long)L * 1600) return;
    int l = (int)(i / 1600), c = (int)(i % 1600);
    int h = c / 400, d = c % 400;
    float v = num[(long)h * L * 400 + (long)l * 400 + d] / z[l * 4 + h];
    o[i] = f2bf(v);
}

// ---------------------------------------------------------------------------
// Host orchestration
// ---------------------------------------------------------------------------
extern "C" void kernel_launch(void* const* d_in, const int* in_sizes, int n_in,
                              void* d_out, int out_size, void* d_ws, size_t ws_size,
                              hipStream_t stream)
{
    (void)in_sizes; (void)n_in; (void)out_size; (void)ws_size;

    // ---- parameter pointers (JAX pytree flatten order: sorted dict keys) ----
    const float* seq  = (const float*)d_in[0];
    const float* atac = (const float*)d_in[1];
    int idx = 2;
    auto nx = [&]() { return (const float*)d_in[idx++]; };

    const float* final_b    = nx();
    const float* final_bn_b = nx();
    const float* final_bn_g = nx();
    const float* final_bn_m = nx();
    const float* final_bn_v = nx();
    const float* final_k    = nx();
    const float* head_b     = nx();
    const float* head_w     = nx();

    struct Layer { const float *b1,*b2,*bk,*bo,*bq,*bv,*ln1_b,*ln1_g,*ln2_b,*ln2_g,*proj,*w1,*w2,*wk,*wo,*wq,*wv; };
    Layer Ls[6];
    for (int l = 0; l < 6; ++l) {
        Ls[l].b1 = nx(); Ls[l].b2 = nx(); Ls[l].bk = nx(); Ls[l].bo = nx();
        Ls[l].bq = nx(); Ls[l].bv = nx();
        Ls[l].ln1_b = nx(); Ls[l].ln1_g = nx(); Ls[l].ln2_b = nx(); Ls[l].ln2_g = nx();
        Ls[l].proj = nx();
        Ls[l].w1 = nx(); Ls[l].w2 = nx();
        Ls[l].wk = nx(); Ls[l].wo = nx(); Ls[l].wq = nx(); Ls[l].wv = nx();
    }
    const float* lnf_b = nx();
    const float* lnf_g = nx();
    const float* stem_at_b = nx();
    const float* stem_at_k = nx();
    const float* stem_b = nx();
    const float* stem_k = nx();
    const float* stem_pool = nx();
    const float* stem_pool_at = nx();

    struct CB { const float *b, *bn_b, *bn_g, *bn_m, *bn_v, *k; };
    auto rdcb = [&]() { CB c; c.b = nx(); c.bn_b = nx(); c.bn_g = nx(); c.bn_m = nx(); c.bn_v = nx(); c.k = nx(); return c; };
    CB stem_res    = rdcb();
    CB stem_res_at = rdcb();
    struct TB { CB c1, c2; const float* pool; };
    TB T[6], TA[2];
    for (int i = 0; i < 6; ++i) { T[i].c1 = rdcb(); T[i].c2 = rdcb(); T[i].pool = nx(); }
    for (int i = 0; i < 2; ++i) { TA[i].c1 = rdcb(); TA[i].c2 = rdcb(); TA[i].pool = nx(); }

    // ---- workspace arena ----
    size_t off = 0;
    auto aU16 = [&](size_t e) { unsigned short* p = (unsigned short*)((char*)d_ws + off);
                                off = (off + e * 2 + 255) & ~(size_t)255; return p; };
    auto aF32 = [&](size_t e) { float* p = (float*)((char*)d_ws + off);
                                off = (off + e * 4 + 255) & ~(size_t)255; return p; };

    unsigned short* wslot  = aU16(10500000);                 // reusable bf16 weight slot (Bt layout)
    unsigned short* projT  = aU16((size_t)256 * 400);
    unsigned short* seq_bf = aU16((size_t)98304 * 4);
    unsigned short* at_bf  = aU16(24576);
    const size_t BIGE = (size_t)98304 * 768;
    unsigned short* bufA = aU16(BIGE);
    unsigned short* bufB = aU16(BIGE);
    unsigned short* bufC = aU16(BIGE);
    unsigned short* aA = aU16((size_t)24576 * 64);
    unsigned short* aB = aU16((size_t)24576 * 64);
    unsigned short* aC = aU16((size_t)24576 * 64);
    unsigned short* xs   = aU16((size_t)768 * 1600);
    unsigned short* xs2  = aU16((size_t)768 * 1600);
    unsigned short* lnb  = aU16((size_t)768 * 1600);
    unsigned short* qh   = aU16((size_t)4 * 768 * 400);
    unsigned short* kh   = aU16((size_t)4 * 768 * 400);
    unsigned short* vhT  = aU16((size_t)4 * 400 * 768);      // Bt layout per head
    unsigned short* qpb  = aU16((size_t)4 * 768 * 256);
    unsigned short* kpb  = aU16((size_t)4 * 768 * 256);
    unsigned short* kvT  = aU16((size_t)4 * 400 * 256);      // Bt layout per head
    unsigned short* obuf = aU16((size_t)768 * 1600);
    unsigned short* h1   = aU16((size_t)768 * 3200);
    float* qf   = aF32((size_t)768 * 1600);
    float* kf   = aF32((size_t)768 * 1600);
    float* vf   = aF32((size_t)768 * 1600);
    float* qpf  = aF32((size_t)4 * 768 * 256);
    float* kpf  = aF32((size_t)4 * 768 * 256);
    float* numf = aF32((size_t)4 * 768 * 400);
    float* ksf  = aF32(4 * 256);
    float* zf   = aF32(768 * 4);

    auto EW = [&](long n) { return dim3((unsigned)((n + 255) / 256)); };
    auto cvt = [&](const float* s, unsigned short* d, long n) {
        f2bf_kernel<<<EW(n), 256, 0, stream>>>(s, d, n);
    };

    // dispatch: fast (aligned) vs guarded (templated by addressing mode)
    auto gemmAny = [&](const unsigned short* A, const unsigned short* Bt,
                       const float* bias, const unsigned short* resid,
                       unsigned short* ob, float* of,
                       int M, int N, int K, int lda, int Ci, int Wk, int dil, int padl,
                       int ta, int row0, int Ltot, int epi, float es, float ea, int otrans) {
        bool conv = (Wk > 1) || (padl != 0);
        bool fast = !ta && !otrans && row0 == 0 &&
                    (M % 64 == 0) && (N % 64 == 0) && (K % 32 == 0) &&
                    (!conv || (Ci % 32 == 0));
        if (fast) {
            dim3 grid(N / 64, M / 64);
            if (conv)
                wmma_gemm_fast<1><<<grid, 256, 0, stream>>>(A, Bt, bias, resid, ob, of,
                    M, N, K, lda, Ci, dil, padl, Ltot, epi, es, ea);
            else
                wmma_gemm_fast<0><<<grid, 256, 0, stream>>>(A, Bt, bias, resid, ob, of,
                    M, N, K, lda, Ci, dil, padl, Ltot, epi, es, ea);
        } else {
            dim3 grid((N + 63) / 64, (M + 31) / 32);
            if (ta)
                wmma_gemm_gen<2><<<grid, 256, 0, stream>>>(A, Bt, nullptr, nullptr, ob, nullptr,
                    M, N, K, lda, 0, 0, 0, 0, 0, epi, es, ea);
            else if (conv)
                wmma_gemm_gen<1><<<grid, 256, 0, stream>>>(A, Bt, bias, resid, ob, of,
                    M, N, K, lda, __builtin_ctz((unsigned)Ci), dil, padl, row0, Ltot, epi, es, ea);
            else
                wmma_gemm_gen<0><<<grid, 256, 0, stream>>>(A, Bt, bias, resid, ob, of,
                    M, N, K, lda, 0, 0, 0, row0, Ltot, epi, es, ea);
        }
    };
    // GEMM with f32 weight W[k, n] -> transposed bf16 into wslot first
    auto gemmW = [&](const unsigned short* A, const float* W,
                     const float* bias, const unsigned short* resid,
                     unsigned short* ob, float* of,
                     int M, int N, int K, int lda, int Ci, int Wk, int dil, int padl,
                     int row0, int Ltot, int epi, float es, float ea) {
        transpose_f2bf_kernel<<<EW((long)K * N), 256, 0, stream>>>(W, wslot, K, N);
        gemmAny(A, wslot, bias, resid, ob, of, M, N, K, lda, Ci, Wk, dil, padl,
                0, row0, Ltot, epi, es, ea, 0);
    };
    auto bngelu = [&](const unsigned short* x, const CB& c, unsigned short* y, long n, int C) {
        bn_gelu_kernel<<<EW(n), 256, 0, stream>>>(x, c.bn_b, c.bn_g, c.bn_m, c.bn_v, y, n, C);
    };

    // ---- input converts ----
    cvt(seq, seq_bf, (long)98304 * 4);
    cvt(atac, at_bf, 24576);

    // ========================= sequence stem =========================
    gemmW(seq_bf, stem_k, stem_b, nullptr, bufA, nullptr,
          98304, 768, 60, 4, 4, 15, 1, 7, 0, 98304, 0, 0, 0);          // guarded conv (Ci=4)
    bngelu(bufA, stem_res, bufB, (long)98304 * 768, 768);
    gemmW(bufB, stem_res.k, stem_res.b, bufA, bufC, nullptr,
          98304, 768, 768, 768, 768, 1, 1, 0, 0, 98304, 0, 0, 0);      // fast (TDM)
    gemmW(bufC, stem_pool, nullptr, nullptr, bufA, nullptr,
          98304, 768, 768, 768, 768, 1, 1, 0, 0, 98304, 0, 0, 0);      // fast (TDM)
    smpool_kernel<<<EW((long)49152 * 768), 256, 0, stream>>>(bufC, bufA, bufB,
        (long)49152 * 768, 768, 2);

    // ========================= conv tower =========================
    unsigned short *cur = bufB, *f1 = bufA, *f2 = bufC;
    int Lx = 49152, ci = 768;
    const int Fs[6] = {768, 896, 1024, 1152, 1280, 1536};
    for (int b = 0; b < 6; ++b) {
        int f = Fs[b];
        bngelu(cur, T[b].c1, f1, (long)Lx * ci, ci);
        gemmW(f1, T[b].c1.k, T[b].c1.b, nullptr, f2, nullptr,           // fast conv
              Lx, f, 5 * ci, ci, ci, 5, 1, 2, 0, Lx, 0, 0, 0);
        bngelu(f2, T[b].c2, cur, (long)Lx * f, f);
        gemmW(cur, T[b].c2.k, T[b].c2.b, f2, f1, nullptr,               // fast (TDM)
              Lx, f, f, f, f, 1, 1, 0, 0, Lx, 0, 0, 0);
        gemmW(f1, T[b].pool, nullptr, nullptr, f2, nullptr,             // fast (TDM)
              Lx, f, f, f, f, 1, 1, 0, 0, Lx, 0, 0, 0);
        smpool_kernel<<<EW((long)(Lx / 2) * f), 256, 0, stream>>>(f1, f2, cur,
            (long)(Lx / 2) * f, f, 2);
        Lx /= 2; ci = f;
    }   // cur = [768, 1536]

    // ========================= ATAC branch =========================
    gemmW(at_bf, stem_at_k, stem_at_b, nullptr, aA, nullptr,
          24576, 16, 15, 1, 1, 15, 1, 7, 0, 24576, 0, 0, 0);           // guarded conv (Ci=1)
    bngelu(aA, stem_res_at, aB, (long)24576 * 16, 16);
    gemmW(aB, stem_res_at.k, stem_res_at.b, aA, aC, nullptr,
          24576, 16, 16, 16, 16, 1, 1, 0, 0, 24576, 0, 0, 0);          // guarded plain (N=16)
    gemmW(aC, stem_pool_at, nullptr, nullptr, aA, nullptr,
          24576, 16, 16, 16, 16, 1, 1, 0, 0, 24576, 0, 0, 0);
    smpool_kernel<<<EW((long)12288 * 16), 256, 0, stream>>>(aC, aA, aB,
        (long)12288 * 16, 16, 2);
    unsigned short* cura = aB;
    int La = 12288; ci = 16;
    const int Fa[2] = {32, 64};
    for (int b = 0; b < 2; ++b) {
        int f = Fa[b];
        bngelu(cura, TA[b].c1, aA, (long)La * ci, ci);
        gemmW(aA, TA[b].c1.k, TA[b].c1.b, nullptr, aC, nullptr,
              La, f, 5 * ci, ci, ci, 5, 2, 4, 0, La, 0, 0, 0);         // guarded conv (Ci=16) / fast (Ci=32)
        bngelu(aC, TA[b].c2, cura, (long)La * f, f);
        gemmW(cura, TA[b].c2.k, TA[b].c2.b, aC, aA, nullptr,
              La, f, f, f, f, 1, 1, 0, 0, La, 0, 0, 0);
        gemmW(aA, TA[b].pool, nullptr, nullptr, aC, nullptr,
              La, f, f, f, f, 1, 1, 0, 0, La, 0, 0, 0);
        smpool_kernel<<<EW((long)(La / 4) * f), 256, 0, stream>>>(aA, aC, cura,
            (long)(La / 4) * f, f, 4);
        La /= 4; ci = f;
    }   // cura = [768, 64]

    // ========================= concat + PE =========================
    concat_kernel<<<EW(768L * 1600), 256, 0, stream>>>(cur, cura, xs, 768);
    addpe_kernel<<<EW(768L * 1600), 256, 0, stream>>>(xs, 768, 1600);

    // ========================= transformer =========================
    const float ratio = 0.0625f;   // 1/sqrt(256)
    for (int l = 0; l < 6; ++l) {
        const Layer& LP = Ls[l];
        layernorm_kernel<<<768, 256, 0, stream>>>(xs, LP.ln1_g, LP.ln1_b, lnb, 1600, 1e-3f);
        gemmW(lnb, LP.wq, LP.bq, nullptr, nullptr, qf,
              768, 1600, 1600, 1600, 1600, 1, 1, 0, 0, 768, 0, 0, 0);   // fast (TDM)
        gemmW(lnb, LP.wk, LP.bk, nullptr, nullptr, kf,
              768, 1600, 1600, 1600, 1600, 1, 1, 0, 0, 768, 0, 0, 0);
        gemmW(lnb, LP.wv, LP.bv, nullptr, nullptr, vf,
              768, 1600, 1600, 1600, 1600, 1, 1, 0, 0, 768, 0, 0, 0);
        rope_split_kernel<<<EW(768L * 1600), 256, 0, stream>>>(qf, qh, 768, 1);
        rope_split_kernel<<<EW(768L * 1600), 256, 0, stream>>>(kf, kh, 768, 1);
        rope_split_kernel<<<EW(768L * 1600), 256, 0, stream>>>(vf, vhT, 768, 2);  // Bt layout
        cvt(LP.proj, projT, (long)256 * 400);                 // proj[m][d] is already Bt[n=m][k=d]
        for (int h = 0; h < 4; ++h) {
            size_t oq = (size_t)h * 768 * 256;
            gemmAny(qh + (size_t)h * 768 * 400, projT, nullptr, nullptr,
                    qpb + oq, qpf + oq, 768, 256, 400, 400,
                    400, 1, 1, 0, 0, 0, 768, 4, ratio, 0.001f, 0);      // guarded plain (K=400)
            gemmAny(kh + (size_t)h * 768 * 400, projT, nullptr, nullptr,
                    kpb + oq, kpf + oq, 768, 256, 400, 400,
                    400, 1, 1, 0, 0, 0, 768, 4, ratio, 0.001f, 0);
        }
        ksum_kernel<<<4, 256, 0, stream>>>(kpf, ksf, 768);
        for (int h = 0; h < 4; ++h) {
            // kvT[h][d][m] = sum_l kp[h][l][m] * v[h][l][d]   (MODE 2: ta + transposed out)
            gemmAny(kpb + (size_t)h * 768 * 256, vhT + (size_t)h * 400 * 768,
                    nullptr, nullptr, kvT + (size_t)h * 400 * 256, nullptr,
                    256, 400, 768, 256, 256, 1, 1, 0, 1, 0, 0, 0, 0, 0, 1);
            // num[h] = qp[h] (768x256) @ kv[h];  B = kvT in Bt layout (K=256)
            gemmAny(qpb + (size_t)h * 768 * 256, kvT + (size_t)h * 400 * 256,
                    nullptr, nullptr, nullptr, numf + (size_t)h * 768 * 400,
                    768, 400, 256, 256, 256, 1, 1, 0, 0, 0, 0, 0, 0, 0, 0);
        }
        zden_kernel<<<EW(768 * 4), 256, 0, stream>>>(qpf, ksf, zf, 768);
        combine_kernel<<<EW(768L * 1600), 256, 0, stream>>>(numf, zf, obuf, 768);
        gemmW(obuf, LP.wo, LP.bo, xs, xs2, nullptr,
              768, 1600, 1600, 1600, 1600, 1, 1, 0, 0, 768, 0, 0, 0);   // fast + resid
        // FFN
        layernorm_kernel<<<768, 256, 0, stream>>>(xs2, LP.ln2_g, LP.ln2_b, lnb, 1600, 1e-3f);
        gemmW(lnb, LP.w1, LP.b1, nullptr, h1, nullptr,
              768, 3200, 1600, 1600, 1600, 1, 1, 0, 0, 768, 1, 0, 0);   // fast, relu
        gemmW(h1, LP.w2, LP.b2, xs2, xs, nullptr,
              768, 1600, 3200, 3200, 3200, 1, 1, 0, 0, 768, 0, 0, 0);   // fast + resid
    }

    // ========================= head =========================
    layernorm_kernel<<<768, 256, 0, stream>>>(xs, lnf_g, lnf_b, lnb, 1600, 1e-3f);
    bn_gelu_kernel<<<EW(768L * 1600), 256, 0, stream>>>(lnb, final_bn_b, final_bn_g,
        final_bn_m, final_bn_v, obuf, 768L * 1600, 1600);
    gemmW(obuf, final_k, final_b, nullptr, h1, nullptr,
          768, 3200, 1600, 1600, 1600, 1, 1, 0, 0, 768, 2, 0, 0);       // fast, gelu
    // trim rows [160, 608), project to 1 channel, softplus -> d_out (f32, 448)
    gemmW(h1, head_w, head_b, nullptr, nullptr, (float*)d_out,
          448, 1, 3200, 3200, 3200, 1, 1, 0, 160, 768, 3, 0, 0);        // guarded plain, row0
}
